// SecurityAwareGNN_43473658970339
// MI455X (gfx1250) — compile-verified, matrix operand order checked by reading, not run on hardware
//
#include <hip/hip_runtime.h>
#include <stdint.h>

typedef unsigned int u32;
typedef unsigned short u16;
typedef float v8f __attribute__((ext_vector_type(8)));
typedef __bf16 v16bf __attribute__((ext_vector_type(16)));

union BFrag {
  v16bf v;
  u32   u[8];
  uint4 q[2];
};

// round-to-nearest-even fp32 -> bf16 pair (used only in one-time weight packing)
__device__ __forceinline__ u32 pack2bf_rn(float x, float y) {
  u32 a = __float_as_uint(x), b = __float_as_uint(y);
  a = (a + 0x7fffu + ((a >> 16) & 1u)) >> 16;
  b = (b + 0x7fffu + ((b >> 16) & 1u)) & 0xffff0000u;
  return a | b;
}
// fast truncating fp32 -> bf16 pair (hot activation paths): 2 VALU ops
__device__ __forceinline__ u32 pack2bf(float x, float y) {
  u32 a = __float_as_uint(x), b = __float_as_uint(y);
  return (a >> 16) | (b & 0xffff0000u);
}
__device__ __forceinline__ u16 f2bf(float x) {
  return (u16)(__float_as_uint(x) >> 16);
}

__device__ __forceinline__ v8f wmma_bf16(const BFrag& a, const BFrag& b, v8f c) {
  return __builtin_amdgcn_wmma_f32_16x16x32_bf16(false, a.v, false, b.v, (short)0, c,
                                                 false, false);
}

// B fragment: lane l -> col n=l&15, half=l>>4; 16 bf16 = W[k][n] for
// k = ks*32 + half*8 + 0..7  then  k = ks*32 + 16 + half*8 + 0..7
__device__ __forceinline__ void load_bfrag(BFrag& f, const uint4* fragbuf, int ks, int nt,
                                           int NT, int lane) {
  const uint4* p = fragbuf + ((size_t)((ks * NT + nt) * 32 + lane)) * 2;
  f.q[0] = p[0];
  f.q[1] = p[1];
}

// ---------------- weight fragment packer (fp32 row-major K x N -> bf16 frags) -------------
__global__ void convert_frag(const float* __restrict__ W, u32* __restrict__ dst, int K, int N) {
  int NT = N >> 4;
  int total = (K * N) >> 1;  // u32 words
  for (int i = blockIdx.x * blockDim.x + threadIdx.x; i < total; i += gridDim.x * blockDim.x) {
    int w = i & 7;
    int lane = (i >> 3) & 31;
    int fid = i >> 8;
    int nt = fid % NT, ks = fid / NT;
    int half = lane >> 4;
    int n = nt * 16 + (lane & 15);
    int j0 = w * 2, j1 = j0 + 1;
    int k0 = ks * 32 + ((j0 < 8) ? (half * 8 + j0) : (16 + half * 8 + (j0 - 8)));
    int k1 = ks * 32 + ((j1 < 8) ? (half * 8 + j1) : (16 + half * 8 + (j1 - 8)));
    dst[i] = pack2bf_rn(W[(size_t)k0 * N + n], W[(size_t)k1 * N + n]);
  }
}

__global__ void zero_f32(float* __restrict__ p, size_t n) {
  for (size_t i = blockIdx.x * (size_t)blockDim.x + threadIdx.x; i < n;
       i += (size_t)gridDim.x * blockDim.x)
    p[i] = 0.0f;
}

// ---------------- node embedding: x[N,32] @ W_node[32,128] + b  (1 WMMA / tile) ----------
__global__ __launch_bounds__(256) void embed_kernel(const float* __restrict__ x_in,
                                                    const uint4* __restrict__ Wf,
                                                    const float* __restrict__ bias,
                                                    float* __restrict__ x_cur,
                                                    u32* __restrict__ xb, int Nn) {
  const int tid = threadIdx.x, lane = tid & 31, wv = tid >> 5;  // wv = n-tile 0..7
  const int half = lane >> 4, nn = lane & 15;
  const int col = wv * 16 + nn;
  const float bv = bias[col];
  BFrag B;
  load_bfrag(B, Wf, 0, wv, 8, lane);
  const int ntiles = Nn >> 4;
  for (int tile = blockIdx.x; tile < ntiles; tile += gridDim.x) {
    BFrag A;
    const float4* xq = (const float4*)(x_in + (size_t)(tile * 16 + nn) * 32);
    float4 c0a = xq[half * 2], c0b = xq[half * 2 + 1];
    float4 c1a = xq[4 + half * 2], c1b = xq[4 + half * 2 + 1];
    A.u[0] = pack2bf(c0a.x, c0a.y); A.u[1] = pack2bf(c0a.z, c0a.w);
    A.u[2] = pack2bf(c0b.x, c0b.y); A.u[3] = pack2bf(c0b.z, c0b.w);
    A.u[4] = pack2bf(c1a.x, c1a.y); A.u[5] = pack2bf(c1a.z, c1a.w);
    A.u[6] = pack2bf(c1b.x, c1b.y); A.u[7] = pack2bf(c1b.z, c1b.w);
    v8f C = {};
    C = wmma_bf16(A, B, C);
#pragma unroll
    for (int v = 0; v < 8; ++v) {
      size_t row = (size_t)(tile * 16 + v + 8 * half);
      float val = C[v] + bv;
      x_cur[row * 128 + col] = val;
      ((u16*)xb)[row * 128 + col] = f2bf(val);
    }
  }
}

// ---------------- attention MLP: sigmoid(relu([xi,xj]@W1+b1)@w2+b2) per edge -------------
__global__ __launch_bounds__(128) void att_kernel(const u32* __restrict__ xb,
                                                  const int* __restrict__ eidx,
                                                  const uint4* __restrict__ W1f,
                                                  const float* __restrict__ b1,
                                                  const float* __restrict__ w2,
                                                  const float* __restrict__ b2p,
                                                  float* __restrict__ attv, int Etot) {
  __shared__ u32 aiU[16 * 128];  // 16 x 256 bf16
  __shared__ float attAcc[16];
  const int tid = threadIdx.x, lane = tid & 31, wv = tid >> 5;  // wv 0..3
  const int half = lane >> 4, nn = lane & 15;
  const int col = wv * 16 + nn;
  const float b1v = b1[col];
  const float w2v = w2[col];
  const float b2s = b2p[0];
  const int ntiles = Etot >> 4;
  for (int tile = blockIdx.x; tile < ntiles; tile += gridDim.x) {
    if (tid < 16) attAcc[tid] = 0.0f;
    for (int idx = tid; idx < 16 * 32; idx += 128) {
      int r = idx >> 5, cg = idx & 31;
      int e = tile * 16 + r;
      int node = (cg < 16) ? eidx[Etot + e] : eidx[e];  // xi = x[dst], xj = x[src]
      ((uint4*)(aiU + r * 128))[cg] = ((const uint4*)(xb + (size_t)node * 64))[cg & 15];
    }
    __syncthreads();
    v8f C = {};
    const uint4* rowp = (const uint4*)(aiU + nn * 128);
#pragma unroll
    for (int ks = 0; ks < 8; ++ks) {
      BFrag A, B;
      A.q[0] = rowp[ks * 4 + half];
      A.q[1] = rowp[ks * 4 + 2 + half];
      load_bfrag(B, W1f, ks, wv, 4, lane);
      C = wmma_bf16(A, B, C);
    }
#pragma unroll
    for (int v = 0; v < 8; ++v) {
      float p = fmaxf(C[v] + b1v, 0.0f) * w2v;
      p += __shfl_xor(p, 1, 32);
      p += __shfl_xor(p, 2, 32);
      p += __shfl_xor(p, 4, 32);
      p += __shfl_xor(p, 8, 32);
      if (nn == 0) atomicAdd(&attAcc[v + 8 * half], p);
    }
    __syncthreads();
    if (tid < 16) {
      float z = attAcc[tid] + b2s;
      attv[tile * 16 + tid] = 1.0f / (1.0f + __expf(-z));
    }
    __syncthreads();
  }
}

// ---------------- fused edge MLP (leg: path=0, eav: path=1) ------------------------------
__global__ __launch_bounds__(256) void edge_mlp_kernel(
    const u32* __restrict__ xb, const float* __restrict__ edge_attr,
    const int* __restrict__ eidx, const int* __restrict__ etype,
    const float* __restrict__ W_edge, const float* __restrict__ b_edge,
    const uint4* __restrict__ W1f, const float* __restrict__ b1,
    const uint4* __restrict__ W2f, const float* __restrict__ b2,
    const float* __restrict__ attv, float* __restrict__ aggr, int Etot, int path) {
  __shared__ u32 combU[16 * 192];   // 16 x 384 bf16 comb tile
  __shared__ float hidL[16 * 128];  // hidden tile fp32
  __shared__ int dstL[16];
  __shared__ float scaleL[16];

  const int tid = threadIdx.x, lane = tid & 31, wv = tid >> 5;  // wv = n-tile 0..7
  const int half = lane >> 4, nn = lane & 15;
  const int col = wv * 16 + nn;
  const float b1v = b1[col];
  const float b2v = b2[col];
  const uint4* rowp = (const uint4*)(combU + nn * 192);

  const int ntiles = Etot >> 4;
  for (int tile = blockIdx.x; tile < ntiles; tile += gridDim.x) {
    // stage edge meta
    if (tid < 16) {
      int e = tile * 16 + tid;
      float etf = (float)etype[e];
      dstL[tid] = eidx[Etot + e];
      scaleL[tid] = (path == 0) ? (1.0f - etf) : (etf * attv[e]);
    }
    // stage comb = [x[dst] | x[src] | ea@W_edge+b_edge] as bf16
    for (int idx = tid; idx < 16 * 48; idx += 256) {
      int r = idx / 48, cg = idx % 48;
      int e = tile * 16 + r;
      uint4 chunk;
      if (cg < 32) {
        int node = (cg < 16) ? eidx[Etot + e] : eidx[e];
        chunk = ((const uint4*)(xb + (size_t)node * 64))[cg & 15];
      } else {
        int c0 = (cg - 32) * 8;
        float acc[8];
#pragma unroll
        for (int j = 0; j < 8; ++j) acc[j] = b_edge[c0 + j];
        const float* eap = edge_attr + (size_t)e * 8;
#pragma unroll
        for (int k = 0; k < 8; ++k) {
          float av = eap[k];
          const float* wr = W_edge + k * 128 + c0;
#pragma unroll
          for (int j = 0; j < 8; ++j) acc[j] += av * wr[j];
        }
        chunk.x = pack2bf(acc[0], acc[1]); chunk.y = pack2bf(acc[2], acc[3]);
        chunk.z = pack2bf(acc[4], acc[5]); chunk.w = pack2bf(acc[6], acc[7]);
      }
      ((uint4*)(combU + r * 192))[cg] = chunk;
    }
    __syncthreads();

    // GEMM1: comb(16x384) @ W1(384x128), K = 12 steps of 32
    v8f C = {};
#pragma unroll
    for (int ks = 0; ks < 12; ++ks) {
      BFrag A, B;
      A.q[0] = rowp[ks * 4 + half];
      A.q[1] = rowp[ks * 4 + 2 + half];
      load_bfrag(B, W1f, ks, wv, 8, lane);
      C = wmma_bf16(A, B, C);
    }
#pragma unroll
    for (int v = 0; v < 8; ++v)
      hidL[(v + 8 * half) * 128 + col] = fmaxf(C[v] + b1v, 0.0f);
    __syncthreads();

    // GEMM2: hid(16x128) @ W2(128x128), K = 4 steps of 32
    v8f C2 = {};
#pragma unroll
    for (int ks = 0; ks < 4; ++ks) {
      BFrag A, B;
      int c0 = ks * 32 + half * 8;
      const float4* hp0 = (const float4*)(hidL + nn * 128 + c0);
      const float4* hp1 = (const float4*)(hidL + nn * 128 + c0 + 16);
      float4 f0 = hp0[0], f1 = hp0[1], f2 = hp1[0], f3 = hp1[1];
      A.u[0] = pack2bf(f0.x, f0.y); A.u[1] = pack2bf(f0.z, f0.w);
      A.u[2] = pack2bf(f1.x, f1.y); A.u[3] = pack2bf(f1.z, f1.w);
      A.u[4] = pack2bf(f2.x, f2.y); A.u[5] = pack2bf(f2.z, f2.w);
      A.u[6] = pack2bf(f3.x, f3.y); A.u[7] = pack2bf(f3.z, f3.w);
      load_bfrag(B, W2f, ks, wv, 8, lane);
      C2 = wmma_bf16(A, B, C2);
    }
    // gated scatter-add to aggr[dst]
#pragma unroll
    for (int v = 0; v < 8; ++v) {
      int row = v + 8 * half;
      float s = scaleL[row];
      if (s != 0.0f)
        atomicAdd(aggr + (size_t)dstL[row] * 128 + col, (C2[v] + b2v) * s);
    }
    __syncthreads();
  }
}

// ---------------- node update: [x,aggr]@W_upd + b, LN, relu+residual, relu ---------------
__global__ __launch_bounds__(256) void upd_kernel(float* __restrict__ x_cur,
                                                  const float* __restrict__ aggr,
                                                  u32* __restrict__ xb,
                                                  const uint4* __restrict__ Wf,
                                                  const float* __restrict__ bu,
                                                  const float* __restrict__ g,
                                                  const float* __restrict__ bb, int Nn) {
  __shared__ u32 inU[16 * 128];  // 16 x 256 bf16
  __shared__ float hidL[16 * 128];
  __shared__ float muL[16], rsL[16];
  const int tid = threadIdx.x, lane = tid & 31, wv = tid >> 5;
  const int half = lane >> 4, nn = lane & 15;
  const int col = wv * 16 + nn;
  const float buv = bu[col];
  const uint4* rowp = (const uint4*)(inU + nn * 128);
  const int ntiles = Nn >> 4;
  for (int tile = blockIdx.x; tile < ntiles; tile += gridDim.x) {
    for (int idx = tid; idx < 16 * 32; idx += 256) {
      int r = idx >> 5, cg = idx & 31;
      size_t node = (size_t)(tile * 16 + r);
      const float* src = (cg < 16) ? (x_cur + node * 128 + cg * 8)
                                   : (aggr + node * 128 + (cg - 16) * 8);
      float4 a = ((const float4*)src)[0], b4 = ((const float4*)src)[1];
      uint4 ch;
      ch.x = pack2bf(a.x, a.y);  ch.y = pack2bf(a.z, a.w);
      ch.z = pack2bf(b4.x, b4.y); ch.w = pack2bf(b4.z, b4.w);
      ((uint4*)(inU + r * 128))[cg] = ch;
    }
    __syncthreads();
    v8f C = {};
#pragma unroll
    for (int ks = 0; ks < 8; ++ks) {
      BFrag A, B;
      A.q[0] = rowp[ks * 4 + half];
      A.q[1] = rowp[ks * 4 + 2 + half];
      load_bfrag(B, Wf, ks, wv, 8, lane);
      C = wmma_bf16(A, B, C);
    }
#pragma unroll
    for (int v = 0; v < 8; ++v) hidL[(v + 8 * half) * 128 + col] = C[v] + buv;
    __syncthreads();
    if (tid < 16) {
      float s = 0.0f;
      for (int c = 0; c < 128; ++c) s += hidL[tid * 128 + c];
      float mu = s * (1.0f / 128.0f);
      float vv = 0.0f;
      for (int c = 0; c < 128; ++c) {
        float d = hidL[tid * 128 + c] - mu;
        vv += d * d;
      }
      muL[tid] = mu;
      rsL[tid] = rsqrtf(vv * (1.0f / 128.0f) + 1e-5f);
    }
    __syncthreads();
    {
      int row = tid & 15, cg2 = tid >> 4, c0 = cg2 * 8;
      size_t node = (size_t)(tile * 16 + row);
      float xs[8];
#pragma unroll
      for (int j = 0; j < 8; ++j) {
        int c = c0 + j;
        float hv = hidL[row * 128 + c];
        float ln = (hv - muL[row]) * rsL[row] * g[c] + bb[c];
        float h2 = fmaxf(ln, 0.0f) + x_cur[node * 128 + c];
        xs[j] = fmaxf(h2, 0.0f);
      }
      float4 o0, o1;
      o0.x = xs[0]; o0.y = xs[1]; o0.z = xs[2]; o0.w = xs[3];
      o1.x = xs[4]; o1.y = xs[5]; o1.z = xs[6]; o1.w = xs[7];
      ((float4*)(x_cur + node * 128 + c0))[0] = o0;
      ((float4*)(x_cur + node * 128 + c0))[1] = o1;
      uint4 xo;
      xo.x = pack2bf(xs[0], xs[1]); xo.y = pack2bf(xs[2], xs[3]);
      xo.z = pack2bf(xs[4], xs[5]); xo.w = pack2bf(xs[6], xs[7]);
      ((uint4*)xb)[node * 16 + cg2] = xo;
    }
    __syncthreads();
  }
}

// ---------------- pooling, refinement head, output assembly -----------------------------
__global__ __launch_bounds__(256) void pool_kernel(const float* __restrict__ x,
                                                   float* __restrict__ gr, int Nn) {
  int col = threadIdx.x & 127, sub = threadIdx.x >> 7;
  float s = 0.0f;
  for (int r = blockIdx.x * 2 + sub; r < Nn; r += gridDim.x * 2)
    s += x[(size_t)r * 128 + col];
  atomicAdd(&gr[col], s);
}

__global__ __launch_bounds__(128) void ref_kernel(const float* __restrict__ gr,
                                                  const float* __restrict__ W1,
                                                  const float* __restrict__ b1,
                                                  const float* __restrict__ W2,
                                                  const float* __restrict__ b2,
                                                  float* __restrict__ refv, float invN) {
  __shared__ float hl[128];
  int t = threadIdx.x;
  float acc = b1[t];
  for (int k = 0; k < 128; ++k) acc += gr[k] * invN * W1[k * 128 + t];
  hl[t] = fmaxf(acc, 0.0f);
  __syncthreads();
  if (t < 16) {
    float o = b2[t];
    for (int k = 0; k < 128; ++k) o += hl[k] * W2[k * 16 + t];
    refv[t] = o;
  }
}

__global__ void out_kernel(const float* __restrict__ beam, const float* __restrict__ ris,
                           const float* __restrict__ traj, const float* __restrict__ refv,
                           const float* __restrict__ gr, float* __restrict__ out, float invN) {
  int t = threadIdx.x;
  if (t < 64) out[t] = beam[t];
  else if (t < 164) out[t] = ris[t - 64];
  else if (t < 167) out[t] = traj[t - 164] + refv[t - 164];
  else if (t < 295) out[t] = gr[t - 167] * invN;
}

// ========================================================================================
extern "C" void kernel_launch(void* const* d_in, const int* in_sizes, int n_in,
                              void* d_out, int out_size, void* d_ws, size_t ws_size,
                              hipStream_t stream) {
  (void)n_in; (void)out_size; (void)ws_size;
  const float* x_in      = (const float*)d_in[0];
  const float* edge_attr = (const float*)d_in[1];
  const int*   eidx      = (const int*)d_in[2];
  const int*   etype     = (const int*)d_in[3];
  const float* beam      = (const float*)d_in[5];
  const float* ris       = (const float*)d_in[6];
  const float* traj      = (const float*)d_in[7];
  const float* W_node = (const float*)d_in[8];  const float* b_node = (const float*)d_in[9];
  const float* W_edge = (const float*)d_in[10]; const float* b_edge = (const float*)d_in[11];
  const float* W_leg1 = (const float*)d_in[12]; const float* b_leg1 = (const float*)d_in[13];
  const float* W_leg2 = (const float*)d_in[14]; const float* b_leg2 = (const float*)d_in[15];
  const float* W_eav1 = (const float*)d_in[16]; const float* b_eav1 = (const float*)d_in[17];
  const float* W_eav2 = (const float*)d_in[18]; const float* b_eav2 = (const float*)d_in[19];
  const float* W_att1 = (const float*)d_in[20]; const float* b_att1 = (const float*)d_in[21];
  const float* W_att2 = (const float*)d_in[22]; const float* b_att2 = (const float*)d_in[23];
  const float* W_upd  = (const float*)d_in[24]; const float* b_upd  = (const float*)d_in[25];
  const float* ln_g   = (const float*)d_in[26]; const float* ln_b   = (const float*)d_in[27];
  const float* W_ref1 = (const float*)d_in[28]; const float* b_ref1 = (const float*)d_in[29];
  const float* W_ref2 = (const float*)d_in[30]; const float* b_ref2 = (const float*)d_in[31];

  const int N = in_sizes[0] / 32;
  const int E = in_sizes[3];
  const int L = in_sizes[13] / 128;

  char* wsb = (char*)d_ws;
  size_t off = 0;
  auto alloc = [&](size_t bytes) -> void* {
    void* p = wsb + off;
    off = (off + bytes + 255) & ~(size_t)255;
    return p;
  };
  u32*   xb    = (u32*)alloc((size_t)N * 128 * 2);   // bf16 node features (gather copy)
  float* x_cur = (float*)alloc((size_t)N * 128 * 4); // fp32 node features
  float* aggr  = (float*)alloc((size_t)N * 128 * 4);
  float* attb  = (float*)alloc((size_t)E * 4);
  float* grref = (float*)alloc(144 * 4);
  float* gr = grref;
  float* refv = grref + 128;
  uint4* fNode = (uint4*)alloc((size_t)32 * 128 * 2);
  uint4 *fLeg1[8], *fLeg2[8], *fEav1[8], *fEav2[8], *fAtt1[8], *fUpd[8];
  for (int l = 0; l < L; ++l) {
    fLeg1[l] = (uint4*)alloc((size_t)384 * 128 * 2);
    fLeg2[l] = (uint4*)alloc((size_t)128 * 128 * 2);
    fEav1[l] = (uint4*)alloc((size_t)384 * 128 * 2);
    fEav2[l] = (uint4*)alloc((size_t)128 * 128 * 2);
    fAtt1[l] = (uint4*)alloc((size_t)256 * 64 * 2);
    fUpd[l]  = (uint4*)alloc((size_t)256 * 128 * 2);
  }

  convert_frag<<<64, 256, 0, stream>>>(W_node, (u32*)fNode, 32, 128);
  for (int l = 0; l < L; ++l) {
    convert_frag<<<64, 256, 0, stream>>>(W_leg1 + (size_t)l * 384 * 128, (u32*)fLeg1[l], 384, 128);
    convert_frag<<<64, 256, 0, stream>>>(W_leg2 + (size_t)l * 128 * 128, (u32*)fLeg2[l], 128, 128);
    convert_frag<<<64, 256, 0, stream>>>(W_eav1 + (size_t)l * 384 * 128, (u32*)fEav1[l], 384, 128);
    convert_frag<<<64, 256, 0, stream>>>(W_eav2 + (size_t)l * 128 * 128, (u32*)fEav2[l], 128, 128);
    convert_frag<<<64, 256, 0, stream>>>(W_att1 + (size_t)l * 256 * 64, (u32*)fAtt1[l], 256, 64);
    convert_frag<<<64, 256, 0, stream>>>(W_upd + (size_t)l * 256 * 128, (u32*)fUpd[l], 256, 128);
  }

  embed_kernel<<<1024, 256, 0, stream>>>(x_in, fNode, b_node, x_cur, xb, N);

  for (int l = 0; l < L; ++l) {
    zero_f32<<<2048, 256, 0, stream>>>(aggr, (size_t)N * 128);
    att_kernel<<<2048, 128, 0, stream>>>(xb, eidx, fAtt1[l], b_att1 + l * 64,
                                         W_att2 + l * 64, b_att2 + l, attb, E);
    edge_mlp_kernel<<<2048, 256, 0, stream>>>(xb, edge_attr, eidx, etype, W_edge, b_edge,
                                              fLeg1[l], b_leg1 + l * 128, fLeg2[l],
                                              b_leg2 + l * 128, attb, aggr, E, 0);
    edge_mlp_kernel<<<2048, 256, 0, stream>>>(xb, edge_attr, eidx, etype, W_edge, b_edge,
                                              fEav1[l], b_eav1 + l * 128, fEav2[l],
                                              b_eav2 + l * 128, attb, aggr, E, 1);
    upd_kernel<<<1024, 256, 0, stream>>>(x_cur, aggr, xb, fUpd[l], b_upd + l * 128,
                                         ln_g + l * 128, ln_b + l * 128, N);
  }

  zero_f32<<<1, 256, 0, stream>>>(gr, 144);
  pool_kernel<<<256, 256, 0, stream>>>(x_cur, gr, N);
  float invN = 1.0f / (float)N;
  ref_kernel<<<1, 128, 0, stream>>>(gr, W_ref1, b_ref1, W_ref2, b_ref2, refv, invN);
  out_kernel<<<1, 512, 0, stream>>>(beam, ris, traj, refv, gr, (float*)d_out, invN);
}
